// MultitaskSequenceModel_37168646980131
// MI455X (gfx1250) — compile-verified, matrix operand order checked by reading, not run on hardware
//
#include <hip/hip_runtime.h>

// ---------------- problem constants ----------------
#define BB 64
#define SS 512
#define FF 64
#define HH 256
#define GG 1024          // 4*H
#define BHL (BB*HH)      // one layer of state

typedef _Float16 half_t;
typedef _Float16 v16h __attribute__((ext_vector_type(16)));
typedef _Float16 v8h  __attribute__((ext_vector_type(8)));
typedef float    v8f  __attribute__((ext_vector_type(8)));
typedef float    f4   __attribute__((ext_vector_type(4)));
typedef unsigned int       u32;
typedef unsigned long long u64;

__device__ __forceinline__ float sigm(float x) { return 1.0f / (1.0f + __expf(-x)); }

// Async global -> LDS copy of 16 bytes per lane (GLOBAL_LOAD_ASYNC_TO_LDS_B128,
// tracked by ASYNCcnt).
__device__ __forceinline__ void async_b128(u32 lds_off, u64 gaddr) {
  asm volatile("global_load_async_to_lds_b128 %0, %1, off"
               :: "v"(lds_off), "v"(gaddr)
               : "memory");
}

// Load one 16x32 f16 tile from a row-major LDS tile (32 halves per row) into
// the WMMA A/B register layout:
//   lanes 0-15  : row = lane,    K = {0..7, 16..23}
//   lanes 16-31 : row = lane-16, K = {8..15, 24..31}
__device__ __forceinline__ v16h ld_tile16(const half_t* t, int lane) {
  int r  = lane & 15;
  int hi = (lane >> 4) & 1;
  v8h lo = *(const v8h*)(t + r * 32 + hi * 8);
  v8h hc = *(const v8h*)(t + r * 32 + 16 + hi * 8);
  return __builtin_shufflevector(lo, hc, 0,1,2,3,4,5,6,7,8,9,10,11,12,13,14,15);
}

// ---------------- generic multi-operand WMMA GEMM (optionally LSTM-fused) ----
// C[M,N] = sum_p A_p[M,K_p] @ W_p^T + bias0 + bias1
//   A_p : f32 row-major (lda)      W_p : f16 row-major (N, K) row stride ldw
// Block = 256 threads (8 wave32), tile 64(M) x 128(N), K-step 32, double-
// buffered GLOBAL_LOAD_ASYNC_TO_LDS staging (4 async ops / wave / stage).
// FUSE: block's 128 columns are gate-interleaved (4 gates x 32 h-cols); the
// LSTM cell update runs in-kernel via an LDS gate buffer (no g round trip).
template <bool FUSE>
__global__ __launch_bounds__(256) void gemm_wmma_k(
    const float* A0, int lda0, const half_t* W0, int ldw0, int K0,
    const float* A1, int lda1, const half_t* W1, int ldw1, int K1,
    const float* A2, int lda2, const half_t* W2, int ldw2, int K2,
    const float* bias0, const float* bias1,
    float* C, int ldc, int N,
    const float* c_in, float* c_out, float* h_out, float* h_store, int hstride) {
  __shared__ char smem[32768];
  float*  sAf = (float*)smem;                 // [2][64*32] f32 A stages
  half_t* sB  = (half_t*)(smem + 16384);      // [2][128*32] f16 B stages
  float*  sG  = (float*)smem;                 // [64*128] gate buffer (FUSE epilogue)

  const int m0 = blockIdx.x * 64;
  const int n0 = blockIdx.y * 128;
  const int tid = threadIdx.x;
  const int lane = tid & 31;
  const int wave = tid >> 5;
  const int wm = wave & 3;
  const int wn = wave >> 2;
  const int j0 = n0 >> 2;  // FUSE: h-column group base (32 per block)

  // ---- pair bookkeeping (pure scalar selection; no indexed array/scratch) ----
  const int ks0 = A0 ? (K0 >> 5) : 0;
  const int ks1 = A1 ? (K1 >> 5) : 0;
  const int ks2 = A2 ? (K2 >> 5) : 0;
  const int total = ks0 + ks1 + ks2;
  auto kcnt = [&](int p) { return p == 0 ? ks0 : (p == 1 ? ks1 : ks2); };

  // staging coordinates (stage-invariant per thread)
  const int ra = tid >> 2;             // A row 0..63
  const int ka = (tid & 3) * 8;        // A k-group (8 floats)
  const int rbl = tid >> 1;            // B local row 0..127
  int rbg;
  if (FUSE) {
    rbg = ((rbl >> 5) * HH) + j0 + (rbl & 31);   // gate-interleaved W rows
  } else {
    rbg = n0 + rbl;                  // clamp OOB rows: garbage only feeds
    if (rbg > N - 1) rbg = N - 1;    // masked-out output columns (nc >= N)
  }
  const int pb = (tid & 1) * 16;       // B k-part (16 halves)

  // per-pair base addresses (computed once; advance by +128B/+64B per k-step)
  const u64 baA0 = (u64)(A0 + (size_t)(m0 + ra) * lda0 + ka);
  const u64 baA1 = A1 ? (u64)(A1 + (size_t)(m0 + ra) * lda1 + ka) : 0;
  const u64 baA2 = A2 ? (u64)(A2 + (size_t)(m0 + ra) * lda2 + ka) : 0;
  const u64 baB0 = (u64)(W0 + (size_t)rbg * ldw0 + pb);
  const u64 baB1 = W1 ? (u64)(W1 + (size_t)rbg * ldw1 + pb) : 0;
  const u64 baB2 = W2 ? (u64)(W2 + (size_t)rbg * ldw2 + pb) : 0;
  auto baseA = [&](int p) { return p == 0 ? baA0 : (p == 1 ? baA1 : baA2); };
  auto baseB = [&](int p) { return p == 0 ? baB0 : (p == 1 ? baB1 : baB2); };

  const u32 laBase = (u32)(u64)(const void*)&sAf[ra * 32 + ka];
  const u32 lbBase = (u32)(u64)(const void*)&sB[rbl * 32 + pb];
  auto issue = [&](u64 ga, u64 gb, int buf) {
    u32 la = laBase + (u32)buf * 8192;
    async_b128(la, ga);
    async_b128(la + 16, ga + 16);
    u32 lb = lbBase + (u32)buf * 8192;
    async_b128(lb, gb);
    async_b128(lb + 16, gb + 16);
  };

  v8f acc[4];
#pragma unroll
  for (int j = 0; j < 4; ++j)
#pragma unroll
    for (int r = 0; r < 8; ++r) acc[j][r] = 0.0f;

  u64 gA = baseA(0), gB = baseB(0);
  issue(gA, gB, 0);  // prologue: stage step 0 into buffer 0

  int pi = 0, ks = 0, buf = 0;
  for (int s = 0; s < total; ++s) {
    int pin = pi, ksn = ks + 1;
    if (ksn == kcnt(pi)) { pin = pi + 1; ksn = 0; }
    const bool more = (s + 1 < total);
    if (more) {
      if (ksn == 0) { gA = baseA(pin); gB = baseB(pin); }
      else          { gA += 32 * sizeof(float); gB += 32 * sizeof(half_t); }
      issue(gA, gB, buf ^ 1);  // prefetch next stage
    }
    // our 4 async ops of stage s complete (async loads retire in order)
    if (more) { asm volatile("s_wait_asynccnt 4" ::: "memory"); }
    else      { asm volatile("s_wait_asynccnt 0" ::: "memory"); }
    __syncthreads();  // all waves' stage-s data visible in LDS

    // ---- A fragment: f32 LDS tile -> f16 registers ----
    const float* at = &sAf[buf * 2048 + (wm * 16 + (lane & 15)) * 32 + ((lane >> 4) & 1) * 8];
    f4 a0 = *(const f4*)at;
    f4 a1 = *(const f4*)(at + 4);
    f4 a2 = *(const f4*)(at + 16);
    f4 a3 = *(const f4*)(at + 20);
    v16h a;
#pragma unroll
    for (int i = 0; i < 4; ++i) {
      a[i]      = (half_t)a0[i];
      a[4 + i]  = (half_t)a1[i];
      a[8 + i]  = (half_t)a2[i];
      a[12 + i] = (half_t)a3[i];
    }
    // ---- B fragments first, then 4 back-to-back WMMAs ----
    const half_t* bt = &sB[buf * 4096 + (wn * 64) * 32];
    v16h b0 = ld_tile16(bt +    0, lane);
    v16h b1 = ld_tile16(bt +  512, lane);
    v16h b2 = ld_tile16(bt + 1024, lane);
    v16h b3 = ld_tile16(bt + 1536, lane);
    acc[0] = __builtin_amdgcn_wmma_f32_16x16x32_f16(false, a, false, b0, (short)0, acc[0], false, false);
    acc[1] = __builtin_amdgcn_wmma_f32_16x16x32_f16(false, a, false, b1, (short)0, acc[1], false, false);
    acc[2] = __builtin_amdgcn_wmma_f32_16x16x32_f16(false, a, false, b2, (short)0, acc[2], false, false);
    acc[3] = __builtin_amdgcn_wmma_f32_16x16x32_f16(false, a, false, b3, (short)0, acc[3], false, false);

    __syncthreads();  // done reading buf before it is overwritten at step s+2
    pi = pin; ks = ksn; buf ^= 1;
  }

  const int mloc = wm * 16 + ((lane >> 4) & 1) * 8;
  const int nl = lane & 15;
  if (FUSE) {
    // ---- stage gates (+bias) to LDS, then in-kernel LSTM cell update ----
#pragma unroll
    for (int j = 0; j < 4; ++j) {
      int lc = wn * 64 + j * 16 + nl;                    // local col 0..127
      int gcol = ((lc >> 5) * HH) + j0 + (lc & 31);      // global gate col
      float bs = 0.0f;
      if (bias0) bs += bias0[gcol];
      if (bias1) bs += bias1[gcol];
#pragma unroll
      for (int r = 0; r < 8; ++r) sG[(mloc + r) * 128 + lc] = acc[j][r] + bs;
    }
    __syncthreads();
#pragma unroll
    for (int q = 0; q < 8; ++q) {
      int lin = tid + q * 256;     // 0..2047
      int m = lin >> 5;            // batch row 0..63
      int jl = lin & 31;           // local h col
      const float* gr = sG + m * 128;
      float iv = sigm(gr[jl]);
      float fv = sigm(gr[32 + jl]);
      float gv = tanhf(gr[64 + jl]);
      float ov = sigm(gr[96 + jl]);
      int sidx = (m0 + m) * HH + j0 + jl;
      float c = fv * c_in[sidx] + iv * gv;
      float h = ov * tanhf(c);
      c_out[sidx] = c;
      h_out[sidx] = h;
      if (h_store) h_store[(size_t)(m0 + m) * hstride + j0 + jl] = h;
    }
  } else {
    // ---- plain store C (f32) per the 16x16 f32 D layout ----
#pragma unroll
    for (int j = 0; j < 4; ++j) {
      int nc = n0 + wn * 64 + j * 16 + nl;
      if (nc < N) {
        float bs = 0.0f;
        if (bias0) bs += bias0[nc];
        if (bias1) bs += bias1[nc];
#pragma unroll
        for (int r = 0; r < 8; ++r) C[(size_t)(m0 + mloc + r) * ldc + nc] = acc[j][r] + bs;
      }
    }
  }
}

// ---------------- fused attention: d-GEMM + score + softmax + context -------
// One 256-thread block per batch row b.
//   d[n]     = sum_k h[b,k] * W1b[n,k]              (W1b = W1[:, H:], f16)
//   score[s] = sum_h tanh(E[b,s,h] + d[h]) * W2[h]
//   attn     = softmax_s(score)
//   ctx[b,h] = sum_s attn[s] * enc[b,s,h]
__global__ __launch_bounds__(256) void attention_k(
    const float* __restrict__ h, const half_t* __restrict__ W1b, int ldw1,
    const float* __restrict__ W2, const float* __restrict__ E,
    const float* __restrict__ enc, float* __restrict__ ctx) {
  __shared__ float sh_[HH];
  __shared__ float sd[HH];
  __shared__ float sattn[SS];
  __shared__ float red[256];
  const int b = blockIdx.x, t = threadIdx.x;
  sh_[t] = h[b * HH + t];
  __syncthreads();
  // d
  float acc = 0.0f;
  const half_t* wr = W1b + (size_t)t * ldw1;
  for (int k = 0; k < HH; k += 8) {
    v8h wv = *(const v8h*)(wr + k);
#pragma unroll
    for (int q = 0; q < 8; ++q) acc += (float)wv[q] * sh_[k + q];
  }
  sd[t] = acc;
  __syncthreads();
  // scores (one s per wave per iteration)
  const int wave = t >> 5, lane = t & 31;
  for (int s = wave; s < SS; s += 8) {
    const float* e = E + ((size_t)b * SS + s) * HH;
    float a2 = 0.0f;
#pragma unroll
    for (int i = 0; i < 8; ++i) {
      int hh = i * 32 + lane;
      a2 += tanhf(e[hh] + sd[hh]) * W2[hh];
    }
#pragma unroll
    for (int off = 16; off > 0; off >>= 1) a2 += __shfl_down(a2, off, 32);
    if (lane == 0) sattn[s] = a2;
  }
  __syncthreads();
  // softmax over the 512 scores
  float m = fmaxf(sattn[t], sattn[t + 256]);
  red[t] = m;
  __syncthreads();
  for (int o = 128; o > 0; o >>= 1) {
    if (t < o) red[t] = fmaxf(red[t], red[t + o]);
    __syncthreads();
  }
  m = red[0];
  __syncthreads();
  float e0 = __expf(sattn[t] - m), e1 = __expf(sattn[t + 256] - m);
  red[t] = e0 + e1;
  __syncthreads();
  for (int o = 128; o > 0; o >>= 1) {
    if (t < o) red[t] += red[t + o];
    __syncthreads();
  }
  float inv = 1.0f / red[0];
  __syncthreads();
  sattn[t] = e0 * inv;
  sattn[t + 256] = e1 * inv;
  __syncthreads();
  // context
  float cacc = 0.0f;
  const float* ep = enc + (size_t)b * SS * HH + t;
  for (int s = 0; s < SS; ++s) cacc += sattn[s] * ep[(size_t)s * HH];
  ctx[b * HH + t] = cacc;
}

// ---------------- utility kernels ----------------
__global__ void cvt_f32_f16_k(const float* __restrict__ s, half_t* __restrict__ d, int n) {
  int i = blockIdx.x * 256 + threadIdx.x;
  if (i < n) d[i] = (half_t)s[i];
}
__global__ void zero_k(float* p, int n) {
  int i = blockIdx.x * 256 + threadIdx.x;
  if (i < n) p[i] = 0.0f;
}

// ---------------- host orchestration ----------------
extern "C" void kernel_launch(void* const* d_in, const int* in_sizes, int n_in,
                              void* d_out, int out_size, void* d_ws, size_t ws_size,
                              hipStream_t stream) {
  (void)in_sizes; (void)n_in; (void)out_size; (void)ws_size;
  const float* x = (const float*)d_in[0];
  const float* eW[2][4]; const float* rW[2][4]; const float* pW[2][4];
  for (int l = 0; l < 2; ++l)
    for (int i = 0; i < 4; ++i) eW[l][i] = (const float*)d_in[1 + l * 4 + i];
  const float* W1r = (const float*)d_in[9];
  const float* W2r = (const float*)d_in[10];
  for (int l = 0; l < 2; ++l)
    for (int i = 0; i < 4; ++i) rW[l][i] = (const float*)d_in[11 + l * 4 + i];
  const float* Wor = (const float*)d_in[19];
  const float* bor = (const float*)d_in[20];
  const float* W1p = (const float*)d_in[21];
  const float* W2p = (const float*)d_in[22];
  for (int l = 0; l < 2; ++l)
    for (int i = 0; i < 4; ++i) pW[l][i] = (const float*)d_in[23 + l * 4 + i];
  const float* Wop = (const float*)d_in[31];
  const float* bop = (const float*)d_in[32];
  float* out = (float*)d_out;

  // ---- workspace bump allocator ----
  char* wp = (char*)d_ws;
  auto alloc_f = [&](size_t n) { float* r = (float*)wp; wp += ((n * 4 + 255) / 256) * 256; return r; };
  auto alloc_h = [&](size_t n) { half_t* r = (half_t*)wp; wp += ((n * 2 + 255) / 256) * 256; return r; };

  float* enc_outs = alloc_f((size_t)BB * SS * HH);  // (B,S,H)
  float* E        = alloc_f((size_t)BB * SS * HH);  // enc_outs @ W1[:, :H]^T
  float* ctx      = alloc_f((size_t)BB * HH);
  // ping-pong LSTM states (fused update writes while siblings read)
  float* hEa = alloc_f((size_t)2 * BHL); float* cEa = alloc_f((size_t)2 * BHL);
  float* hEb = alloc_f((size_t)2 * BHL); float* cEb = alloc_f((size_t)2 * BHL);
  float* hDa = alloc_f((size_t)2 * BHL); float* cDa = alloc_f((size_t)2 * BHL);
  float* hDb = alloc_f((size_t)2 * BHL); float* cDb = alloc_f((size_t)2 * BHL);

  half_t* eWih16[2] = { alloc_h((size_t)GG * FF), alloc_h((size_t)GG * HH) };
  half_t* eWhh16[2] = { alloc_h((size_t)GG * HH), alloc_h((size_t)GG * HH) };
  half_t* W1r16 = alloc_h((size_t)HH * 2 * HH);
  half_t* rWih16[2] = { alloc_h((size_t)GG * 2 * HH), alloc_h((size_t)GG * HH) };
  half_t* rWhh16[2] = { alloc_h((size_t)GG * HH), alloc_h((size_t)GG * HH) };
  half_t* Wor16 = alloc_h((size_t)FF * HH);
  half_t* W1p16 = alloc_h((size_t)HH * 2 * HH);
  half_t* pWih16[2] = { alloc_h((size_t)GG * 2 * HH), alloc_h((size_t)GG * HH) };
  half_t* pWhh16[2] = { alloc_h((size_t)GG * HH), alloc_h((size_t)GG * HH) };
  half_t* Wop16 = alloc_h((size_t)FF * HH);

  auto cvt = [&](const float* s, half_t* d, int n) {
    cvt_f32_f16_k<<<(n + 255) / 256, 256, 0, stream>>>(s, d, n);
  };
  cvt(eW[0][0], eWih16[0], GG * FF);  cvt(eW[0][1], eWhh16[0], GG * HH);
  cvt(eW[1][0], eWih16[1], GG * HH);  cvt(eW[1][1], eWhh16[1], GG * HH);
  cvt(W1r, W1r16, HH * 2 * HH);
  cvt(rW[0][0], rWih16[0], GG * 2 * HH); cvt(rW[0][1], rWhh16[0], GG * HH);
  cvt(rW[1][0], rWih16[1], GG * HH);     cvt(rW[1][1], rWhh16[1], GG * HH);
  cvt(Wor, Wor16, FF * HH);
  cvt(W1p, W1p16, HH * 2 * HH);
  cvt(pW[0][0], pWih16[0], GG * 2 * HH); cvt(pW[0][1], pWhh16[0], GG * HH);
  cvt(pW[1][0], pWih16[1], GG * HH);     cvt(pW[1][1], pWhh16[1], GG * HH);
  cvt(Wop, Wop16, FF * HH);

  zero_k<<<(2 * BHL + 255) / 256, 256, 0, stream>>>(hEa, 2 * BHL);
  zero_k<<<(2 * BHL + 255) / 256, 256, 0, stream>>>(cEa, 2 * BHL);

  auto gemmN = [&](const float* A0, int lda0, const half_t* Wm0, int ldw0, int K0,
                   const float* b0, float* C, int ldc, int M, int N) {
    dim3 grid(M / 64, (N + 127) / 128);
    gemm_wmma_k<false><<<grid, 256, 0, stream>>>(
        A0, lda0, Wm0, ldw0, K0,
        nullptr, 0, nullptr, 0, 0, nullptr, 0, nullptr, 0, 0,
        b0, nullptr, C, ldc, N, nullptr, nullptr, nullptr, nullptr, 0);
  };
  auto gemmF = [&](const float* A0, int lda0, const half_t* Wm0, int ldw0, int K0,
                   const float* A1, int lda1, const half_t* Wm1, int ldw1, int K1,
                   const float* A2, int lda2, const half_t* Wm2, int ldw2, int K2,
                   const float* b0, const float* b1,
                   const float* ci, float* co, float* ho, float* hs, int hstride) {
    dim3 grid(1, GG / 128);
    gemm_wmma_k<true><<<grid, 256, 0, stream>>>(
        A0, lda0, Wm0, ldw0, K0, A1, lda1, Wm1, ldw1, K1, A2, lda2, Wm2, ldw2, K2,
        b0, b1, nullptr, 0, GG, ci, co, ho, hs, hstride);
  };

  // ================= encoder (2 fused kernels per step) =================
  for (int t = 0; t < SS; ++t) {
    float* hR = (t & 1) ? hEb : hEa; float* hW = (t & 1) ? hEa : hEb;
    float* cR = (t & 1) ? cEb : cEa; float* cW = (t & 1) ? cEa : cEb;
    gemmF(x + (size_t)t * FF, SS * FF, eWih16[0], FF, FF,
          hR, HH, eWhh16[0], HH, HH,
          nullptr, 0, nullptr, 0, 0,
          eW[0][2], eW[0][3], cR, cW, hW, nullptr, 0);
    gemmF(hW, HH, eWih16[1], HH, HH,
          hR + BHL, HH, eWhh16[1], HH, HH,
          nullptr, 0, nullptr, 0, 0,
          eW[1][2], eW[1][3], cR + BHL, cW + BHL, hW + BHL,
          enc_outs + (size_t)t * HH, SS * HH);
  }
  // final encoder states land in hEa/cEa (t=511 writes the 'a' buffers)
  hipMemcpyAsync(hDa, hEa, (size_t)2 * BHL * sizeof(float), hipMemcpyDeviceToDevice, stream);
  hipMemcpyAsync(cDa, cEa, (size_t)2 * BHL * sizeof(float), hipMemcpyDeviceToDevice, stream);

  // ================= reconstruction decoder (4 kernels per step) =========
  gemmN(enc_outs, HH, W1r16, 2 * HH, HH, nullptr, E, HH, BB * SS, HH);

  for (int t = 0; t < SS; ++t) {
    float* hR = (t & 1) ? hDb : hDa; float* hW = (t & 1) ? hDa : hDb;
    float* cR = (t & 1) ? cDb : cDa; float* cW = (t & 1) ? cDa : cDb;
    attention_k<<<BB, 256, 0, stream>>>(hR + BHL, W1r16 + HH, 2 * HH, W2r, E, enc_outs, ctx);
    gemmF(ctx, HH, rWih16[0], 2 * HH, HH,          // inp = [ctx, xt], xt == h[1]
          hR + BHL, HH, rWih16[0] + HH, 2 * HH, HH,
          hR, HH, rWhh16[0], HH, HH,
          rW[0][2], rW[0][3], cR, cW, hW, nullptr, 0);
    gemmF(hW, HH, rWih16[1], HH, HH,
          hR + BHL, HH, rWhh16[1], HH, HH,
          nullptr, 0, nullptr, 0, 0,
          rW[1][2], rW[1][3], cR + BHL, cW + BHL, hW + BHL, nullptr, 0);
    gemmN(hW + BHL, HH, Wor16, HH, HH, bor,
          out + (size_t)(SS - 1 - t) * FF, SS * FF, BB, FF);
  }

  // ================= prediction head (uses encoder final states) =========
  gemmN(enc_outs, HH, W1p16, 2 * HH, HH, nullptr, E, HH, BB * SS, HH);
  attention_k<<<BB, 256, 0, stream>>>(hEa + BHL, W1p16 + HH, 2 * HH, W2p, E, enc_outs, ctx);
  gemmF(hEa + BHL, HH, pWih16[0], 2 * HH, HH,      // inp = [x0, ctx], x0 == h_n[-1]
        ctx, HH, pWih16[0] + HH, 2 * HH, HH,
        hEa, HH, pWhh16[0], HH, HH,
        pW[0][2], pW[0][3], cEa, cDb, hDb, nullptr, 0);
  gemmF(hDb, HH, pWih16[1], HH, HH,
        hEa + BHL, HH, pWhh16[1], HH, HH,
        nullptr, 0, nullptr, 0, 0,
        pW[1][2], pW[1][3], cEa + BHL, cDb + BHL, hDb + BHL, nullptr, 0);
  gemmN(hDb + BHL, HH, Wop16, HH, HH, bop, out + (size_t)BB * SS * FF, FF, BB, FF);
}
